// SocialLDG_77137612636340
// MI455X (gfx1250) — matrix operand order repaired and checked by reading, not compiled, not run on hardware
//
#include <hip/hip_runtime.h>
#include <cstdint>
#include <cstddef>

// Problem dims (from reference)
#define B_    4096
#define T_    6
#define ZD_   2048
#define HD_   256
#define NH_   4
#define DH_   64
#define I_    1024
#define COUT_ 8
#define BT_   (B_ * T_)      // 24576 rows
#define CD_   (ZD_ + HD_)    // 2304 concat dim
#define FF_   (2 * HD_)      // 512

typedef __bf16 bf16;
typedef __attribute__((ext_vector_type(16))) __bf16 v16bf;
typedef __attribute__((ext_vector_type(8)))  __bf16 v8bf;
typedef __attribute__((ext_vector_type(8)))  float  v8f;
typedef __attribute__((ext_vector_type(4)))  float  v4f;

// ---------------- helpers ----------------

__device__ __forceinline__ bf16 to_bf(float f) {
  unsigned u = __float_as_uint(f);
  u += 0x7FFFu + ((u >> 16) & 1u);              // RNE
  unsigned short hs = (unsigned short)(u >> 16);
  return __builtin_bit_cast(bf16, hs);
}

union ABu { v16bf v; struct { v8bf lo, hi; } h; };

// A tile 16x32 bf16, row-major source (lda elems). ISA 16-bit A layout:
// lane L: m = L&15; k = 8*(L>>4) + (e&7) + 16*(e>>3)
__device__ __forceinline__ v16bf load_a_bf16(const bf16* A, int lda, int m0, int k0, int lane) {
  const bf16* p = A + (size_t)(m0 + (lane & 15)) * lda + k0 + 8 * (lane >> 4);
  ABu r;
  r.h.lo = *(const v8bf*)(p);
  r.h.hi = *(const v8bf*)(p + 16);
  return r.v;
}

// B tile 32x16 from BT stored [N,K] row-major (ldb = K elems).
// lane L: n = L&15; k = 8*(L>>4) + (e&7) + 16*(e>>3)
__device__ __forceinline__ v16bf load_b(const bf16* BT, int ldb, int n0, int k0, int lane) {
  const bf16* p = BT + (size_t)(n0 + (lane & 15)) * ldb + k0 + 8 * (lane >> 4);
  ABu r;
  r.h.lo = *(const v8bf*)(p);
  r.h.hi = *(const v8bf*)(p + 16);
  return r.v;
}

__device__ __forceinline__ v8f wmma_bf(v16bf a, v16bf b, v8f c) {
  return __builtin_amdgcn_wmma_f32_16x16x32_bf16(false, a, false, b, (short)0, c, false, false);
}

#define ZERO8 {0.f,0.f,0.f,0.f,0.f,0.f,0.f,0.f}

// Straight-line epilogue, compile-time mode.
// 0: outf = acc+bias; 1: outb = bf(relu); 2: outb = bf(1+tanh);
// 3: outb = bf(gelu exact); 4: outf += acc+bias
template<int MODE>
__device__ __forceinline__ void epi(v8f acc, int m0, int n0, int lane, int N,
                                    const float* __restrict__ bias,
                                    float* __restrict__ outf, bf16* __restrict__ outb) {
#pragma unroll
  for (int r = 0; r < 8; r++) {
    int m = m0 + r + 8 * (lane >> 4);
    int n = n0 + (lane & 15);
    float v = acc[r] + bias[n];
    if constexpr (MODE == 0)      outf[(size_t)m * N + n] = v;
    else if constexpr (MODE == 1) outb[(size_t)m * N + n] = to_bf(fmaxf(v, 0.f));
    else if constexpr (MODE == 2) outb[(size_t)m * N + n] = to_bf(1.f + tanhf(v));
    else if constexpr (MODE == 3) outb[(size_t)m * N + n] = to_bf(0.5f * v * (1.f + erff(v * 0.70710678118f)));
    else                          outf[(size_t)m * N + n] += v;
  }
}

// ------------- main WMMA GEMM: wave computes 32x64 (2 M-tiles x 4 N-tiles) -------------
// Block 128 threads = 4 waves -> block tile 32 x 256.  M mult of 32, N mult of 256.
// Per K-step: 2 A loads + 4 B loads feed 8 WMMAs (0.75 loads/WMMA).
template<int MODE>
__global__ void k_gemm4(const bf16* __restrict__ A, const bf16* __restrict__ BT,
                        const float* __restrict__ bias, float* __restrict__ outf,
                        bf16* __restrict__ outb, int N, int K, int lda) {
  const int lane = threadIdx.x & 31;
  const int wave = threadIdx.x >> 5;
  const int m0   = blockIdx.x * 32;
  const int n0   = blockIdx.y * 256 + wave * 64;
  v8f c00 = ZERO8, c01 = ZERO8, c02 = ZERO8, c03 = ZERO8;
  v8f c10 = ZERO8, c11 = ZERO8, c12 = ZERO8, c13 = ZERO8;
  for (int k0 = 0; k0 < K; k0 += 32) {
    v16bf a0 = load_a_bf16(A, lda, m0,      k0, lane);
    v16bf a1 = load_a_bf16(A, lda, m0 + 16, k0, lane);
    v16bf b0 = load_b(BT, K, n0,      k0, lane);
    v16bf b1 = load_b(BT, K, n0 + 16, k0, lane);
    v16bf b2 = load_b(BT, K, n0 + 32, k0, lane);
    v16bf b3 = load_b(BT, K, n0 + 48, k0, lane);
    c00 = wmma_bf(a0, b0, c00);
    c01 = wmma_bf(a0, b1, c01);
    c02 = wmma_bf(a0, b2, c02);
    c03 = wmma_bf(a0, b3, c03);
    c10 = wmma_bf(a1, b0, c10);
    c11 = wmma_bf(a1, b1, c11);
    c12 = wmma_bf(a1, b2, c12);
    c13 = wmma_bf(a1, b3, c13);
  }
  epi<MODE>(c00, m0,      n0,      lane, N, bias, outf, outb);
  epi<MODE>(c01, m0,      n0 + 16, lane, N, bias, outf, outb);
  epi<MODE>(c02, m0,      n0 + 32, lane, N, bias, outf, outb);
  epi<MODE>(c03, m0,      n0 + 48, lane, N, bias, outf, outb);
  epi<MODE>(c10, m0 + 16, n0,      lane, N, bias, outf, outb);
  epi<MODE>(c11, m0 + 16, n0 + 16, lane, N, bias, outf, outb);
  epi<MODE>(c12, m0 + 16, n0 + 32, lane, N, bias, outf, outb);
  epi<MODE>(c13, m0 + 16, n0 + 48, lane, N, bias, outf, outb);
}

// edge[b,h,ts] = mod[b, h*I : ] @ QKs[h]   (N=48 padded, write n<36)
__global__ void k_gemm_edge(const bf16* __restrict__ mod, const bf16* __restrict__ qks,
                            float* __restrict__ edge) {
  const int hh = blockIdx.z;
  const int lane = threadIdx.x & 31;
  const int wave = threadIdx.x >> 5;
  const int m0   = blockIdx.x * 16;
  const int n0   = wave * 16;
  if (n0 >= 48) return;
  const bf16* A   = mod + hh * I_;
  const bf16* BTh = qks + (size_t)hh * 48 * I_;
  v8f acc = ZERO8;
#pragma unroll 2
  for (int k0 = 0; k0 < I_; k0 += 32) {
    v16bf a = load_a_bf16(A, NH_ * I_, m0, k0, lane);
    v16bf b = load_b(BTh, I_, n0, k0, lane);
    acc = wmma_bf(a, b, acc);
  }
#pragma unroll
  for (int r = 0; r < 8; r++) {
    int m = m0 + r + 8 * (lane >> 4);
    int n = n0 + (lane & 15);
    if (n < 36) edge[((size_t)m * NH_ + hh) * 36 + n] = acc[r];
  }
}

// head fc1: A = concat(h_bf[:,t,:], z_bf) virtual, K=2304, N=256; per-t weights.
// Wave computes 32x64 like k_gemm4.
__global__ void k_gemm_head1(const bf16* __restrict__ hbf, const bf16* __restrict__ zbf,
                             const bf16* __restrict__ BT, const float* __restrict__ bias,
                             float* __restrict__ fh) {
  const int t = blockIdx.z;
  const int lane = threadIdx.x & 31;
  const int wave = threadIdx.x >> 5;
  const int m0   = blockIdx.x * 32;
  const int n0   = wave * 64;
  const bf16* BTt = BT + (size_t)t * HD_ * CD_;
  v8f c00 = ZERO8, c01 = ZERO8, c02 = ZERO8, c03 = ZERO8;
  v8f c10 = ZERO8, c11 = ZERO8, c12 = ZERO8, c13 = ZERO8;
  for (int k0 = 0; k0 < CD_; k0 += 32) {
    int mA = m0 + (lane & 15);
    int mB = mA + 16;
    int kb = k0 + 8 * (lane >> 4);
    const bf16* pa = (k0 < HD_) ? (hbf + ((size_t)mA * T_ + t) * HD_ + kb)
                                : (zbf + (size_t)mA * ZD_ + (kb - HD_));
    const bf16* pb = (k0 < HD_) ? (hbf + ((size_t)mB * T_ + t) * HD_ + kb)
                                : (zbf + (size_t)mB * ZD_ + (kb - HD_));
    ABu a0, a1;
    a0.h.lo = *(const v8bf*)(pa);
    a0.h.hi = *(const v8bf*)(pa + 16);
    a1.h.lo = *(const v8bf*)(pb);
    a1.h.hi = *(const v8bf*)(pb + 16);
    v16bf b0 = load_b(BTt, CD_, n0,      k0, lane);
    v16bf b1 = load_b(BTt, CD_, n0 + 16, k0, lane);
    v16bf b2 = load_b(BTt, CD_, n0 + 32, k0, lane);
    v16bf b3 = load_b(BTt, CD_, n0 + 48, k0, lane);
    c00 = wmma_bf(a0.v, b0, c00);
    c01 = wmma_bf(a0.v, b1, c01);
    c02 = wmma_bf(a0.v, b2, c02);
    c03 = wmma_bf(a0.v, b3, c03);
    c10 = wmma_bf(a1.v, b0, c10);
    c11 = wmma_bf(a1.v, b1, c11);
    c12 = wmma_bf(a1.v, b2, c12);
    c13 = wmma_bf(a1.v, b3, c13);
  }
  v8f accs[8] = {c00, c01, c02, c03, c10, c11, c12, c13};
#pragma unroll
  for (int q = 0; q < 8; q++) {
    int mt = (q >> 2) * 16;
#pragma unroll
    for (int r = 0; r < 8; r++) {
      int m = m0 + mt + r + 8 * (lane >> 4);
      int n = n0 + (q & 3) * 16 + (lane & 15);
      fh[((size_t)m * T_ + t) * HD_ + n] = accs[q][r] + bias[t * HD_ + n];
    }
  }
}

// ---------------- pack / convert kernels ----------------

// W f32 [K,N] row-major -> BT bf16 [N,K]
__global__ void k_pack_bt(const float* __restrict__ W, bf16* __restrict__ BT, int K, int N) {
  int idx = blockIdx.x * 256 + threadIdx.x;
  if (idx >= K * N) return;
  int k = idx / N, n = idx % N;
  BT[(size_t)n * K + k] = to_bf(W[idx]);
}

// W f32 [Tt,K,N] -> BT bf16 [Tt,N,K]
__global__ void k_pack_bt3(const float* __restrict__ W, bf16* __restrict__ BT, int Tt, int K, int N) {
  int idx = blockIdx.x * 256 + threadIdx.x;
  if (idx >= Tt * K * N) return;
  int t = idx / (K * N), r = idx % (K * N);
  int k = r / N, n = r % N;
  BT[((size_t)t * N + n) * K + k] = to_bf(W[idx]);
}

// elementwise f32 -> bf16, 8 per thread, vectorized (n multiple of 8)
__global__ void k_cvt_bf(const float* __restrict__ X, bf16* __restrict__ Y, int n8) {
  int i = blockIdx.x * 256 + threadIdx.x;
  if (i >= n8) return;
  v4f x0 = *(const v4f*)(X + (size_t)i * 8);
  v4f x1 = *(const v4f*)(X + (size_t)i * 8 + 4);
  v8bf y;
#pragma unroll
  for (int j = 0; j < 4; j++) { y[j] = to_bf(x0[j]); y[4 + j] = to_bf(x1[j]); }
  *(v8bf*)(Y + (size_t)i * 8) = y;
}

// ---------------- small (T=6) kernels ----------------

__global__ void k_small_mm(const float* __restrict__ A, const float* __restrict__ W,
                           const float* __restrict__ bias, float* __restrict__ out,
                           int rows, int K, int N) {
  int idx = blockIdx.x * 256 + threadIdx.x;
  if (idx >= rows * N) return;
  int r = idx / N, n = idx % N;
  float a = bias[n];
  for (int k = 0; k < K; k++) a += A[(size_t)r * K + k] * W[(size_t)k * N + n];
  out[idx] = a;
}

__global__ void k_tpi_finish(const float* __restrict__ ttp1, const float* __restrict__ g,
                             const float* __restrict__ be, const float* __restrict__ p2,
                             const float* __restrict__ b2, float* __restrict__ tpr) {
  __shared__ float r1[256], r2[256], sv[256];
  int row = blockIdx.x, c = threadIdx.x;
  float x = ttp1[row * 256 + c];
  r1[c] = x; r2[c] = x * x; __syncthreads();
  for (int off = 128; off > 0; off >>= 1) {
    if (c < off) { r1[c] += r1[c + off]; r2[c] += r2[c + off]; }
    __syncthreads();
  }
  float mu = r1[0] * (1.f / 256.f);
  float var = r2[0] * (1.f / 256.f) - mu * mu;
  float y = (x - mu) * rsqrtf(var + 1e-5f) * g[c] + be[c];
  sv[c] = fmaxf(y, 0.f);
  __syncthreads();
  float a = b2[c];
  for (int k = 0; k < 256; k++) a += sv[k] * p2[k * 256 + c];
  tpr[row * 256 + c] = a;
}

__global__ void k_qks(const float* __restrict__ Q, const float* __restrict__ K,
                      const float* __restrict__ temp, bf16* __restrict__ qks) {
  int idx = blockIdx.x * 256 + threadIdx.x;
  if (idx >= NH_ * 48 * I_) return;
  int hh = idx / (48 * I_);
  int r  = (idx / I_) % 48;
  int i  = idx % I_;
  float val = 0.f;
  if (r < 36) {
    int t = r / 6, s = r % 6;
    float tc = fmaxf(temp[hh], 0.01f);
    val = Q[(size_t)t * (NH_ * I_) + hh * I_ + i] *
          K[(size_t)s * (NH_ * I_) + hh * I_ + i] * (0.03125f / tc);
  }
  qks[idx] = to_bf(val);
}

__global__ void k_expand(const float* __restrict__ hb, const float* __restrict__ tpr,
                         float* __restrict__ h) {
  int idx = blockIdx.x * 256 + threadIdx.x;
  if (idx >= BT_ * 256) return;
  int n = idx & 255, bt = idx >> 8;
  int t = bt % T_, b = bt / T_;
  h[idx] = hb[(size_t)b * 256 + n] + tpr[t * 256 + n];
}

// ---------------- LayerNorm (wave per row, 256 wide, vectorized) ----------------

__global__ void k_ln_bf16(const float* __restrict__ X, const float* __restrict__ g,
                          const float* __restrict__ be, bf16* __restrict__ Y, int rows) {
  int gid = blockIdx.x * blockDim.x + threadIdx.x;
  int row = gid >> 5, lane = gid & 31;
  if (row >= rows) return;
  const float* x = X + (size_t)row * 256 + lane * 8;
  v4f x0 = *(const v4f*)(x);
  v4f x1 = *(const v4f*)(x + 4);
  float v[8], s = 0.f, s2 = 0.f;
#pragma unroll
  for (int j = 0; j < 4; j++) { v[j] = x0[j]; v[4 + j] = x1[j]; }
#pragma unroll
  for (int j = 0; j < 8; j++) { s += v[j]; s2 += v[j] * v[j]; }
#pragma unroll
  for (int m = 16; m > 0; m >>= 1) { s += __shfl_xor(s, m, 32); s2 += __shfl_xor(s2, m, 32); }
  float mu = s * (1.f / 256.f);
  float var = s2 * (1.f / 256.f) - mu * mu;
  float rs = rsqrtf(var + 1e-5f);
  v8bf y;
#pragma unroll
  for (int j = 0; j < 8; j++) { int n = lane * 8 + j; y[j] = to_bf((v[j] - mu) * rs * g[n] + be[n]); }
  *(v8bf*)(Y + (size_t)row * 256 + lane * 8) = y;
}

// ---------------- attention core (per batch, T=6) ----------------

__global__ void k_attn(const float* __restrict__ q, const float* __restrict__ k,
                       const float* __restrict__ v, const float* __restrict__ edge,
                       float* __restrict__ h, float* __restrict__ alpha0, int step) {
  __shared__ float qs[T_][256], ks[T_][256], vs[T_][256];
  __shared__ float alf[NH_][36];
  int b = blockIdx.x, c = threadIdx.x;
  for (int t = 0; t < T_; t++) {
    size_t idx = ((size_t)b * T_ + t) * 256 + c;
    qs[t][c] = q[idx]; ks[t][c] = k[idx]; vs[t][c] = v[idx];
  }
  __syncthreads();
  if (c < NH_ * 36) {
    int hh = c / 36, r = c % 36, t = r / 6, s = r % 6;
    bool masked = ((t == 0 || t == 4) && (s == 1 || s == 5));
    float d = 0.f;
    for (int dd = 0; dd < DH_; dd++) d += qs[t][hh * DH_ + dd] * ks[s][hh * DH_ + dd];
    float sc = d * 0.125f + edge[((size_t)b * NH_ + hh) * 36 + r];
    alf[hh][r] = masked ? 0.f : 1.f / (1.f + expf(-sc));
  }
  __syncthreads();
  {
    int hh = c >> 6;
    for (int t = 0; t < T_; t++) {
      float a = 0.f;
      for (int s = 0; s < T_; s++) a += alf[hh][t * 6 + s] * vs[s][c];
      h[((size_t)b * T_ + t) * 256 + c] += a;
    }
  }
  if (step == 0 && c < 36)
    alpha0[(size_t)b * 36 + c] = 0.25f * (alf[0][c] + alf[1][c] + alf[2][c] + alf[3][c]);
}

// ---------------- head stage 2: LN+gelu+fc2 (COUT=8) ----------------

__global__ void k_head2(const float* __restrict__ fh, const float* __restrict__ g,
                        const float* __restrict__ be, const float* __restrict__ w2,
                        const float* __restrict__ b2, float* __restrict__ preds) {
  __shared__ float r1[256], r2[256], sv[256];
  int row = blockIdx.x, t = row % T_, c = threadIdx.x;
  float x = fh[(size_t)row * 256 + c];
  r1[c] = x; r2[c] = x * x; __syncthreads();
  for (int off = 128; off > 0; off >>= 1) {
    if (c < off) { r1[c] += r1[c + off]; r2[c] += r2[c + off]; }
    __syncthreads();
  }
  float mu = r1[0] * (1.f / 256.f);
  float var = r2[0] * (1.f / 256.f) - mu * mu;
  float y = (x - mu) * rsqrtf(var + 1e-5f) * g[t * 256 + c] + be[t * 256 + c];
  sv[c] = 0.5f * y * (1.f + erff(y * 0.70710678118f));
  __syncthreads();
  if (c < COUT_) {
    float a = b2[t * COUT_ + c];
    for (int j = 0; j < 256; j++) a += sv[j] * w2[((size_t)t * 256 + j) * COUT_ + c];
    preds[(size_t)row * COUT_ + c] = a;
  }
}

// deterministic fixed-order mean of alpha0
__global__ void k_mean(const float* __restrict__ a0, float* __restrict__ out) {
  __shared__ float r[256];
  int c = threadIdx.x;
  float s = 0.f;
  for (int i = c; i < B_ * 36; i += 256) s += a0[i];
  r[c] = s; __syncthreads();
  for (int off = 128; off > 0; off >>= 1) {
    if (c < off) r[c] += r[c + off];
    __syncthreads();
  }
  if (c == 0) out[0] = r[0] / (float)(B_ * 36);
}

// ---------------- host ----------------

extern "C" void kernel_launch(void* const* d_in, const int* in_sizes, int n_in,
                              void* d_out, int out_size, void* d_ws, size_t ws_size,
                              hipStream_t stream) {
  (void)in_sizes; (void)n_in; (void)out_size; (void)ws_size;
  const float* z           = (const float*)d_in[0];
  const float* tt          = (const float*)d_in[1];
  const float* tpi_zproj_w = (const float*)d_in[2];
  const float* tpi_zproj_b = (const float*)d_in[3];
  const float* tpi_p1_w    = (const float*)d_in[4];
  const float* tpi_p1_b    = (const float*)d_in[5];
  const float* tpi_ln_g    = (const float*)d_in[6];
  const float* tpi_ln_b    = (const float*)d_in[7];
  const float* tpi_p2_w    = (const float*)d_in[8];
  const float* tpi_p2_b    = (const float*)d_in[9];
  const float* eg_q_w      = (const float*)d_in[10];
  const float* eg_q_b      = (const float*)d_in[11];
  const float* eg_k_w      = (const float*)d_in[12];
  const float* eg_k_b      = (const float*)d_in[13];
  const float* eg_c1_w     = (const float*)d_in[14];
  const float* eg_c1_b     = (const float*)d_in[15];
  const float* eg_c2_w     = (const float*)d_in[16];
  const float* eg_c2_b     = (const float*)d_in[17];
  const float* eg_temp     = (const float*)d_in[18];
  const float* wq_w        = (const float*)d_in[19];
  const float* wq_b        = (const float*)d_in[20];
  const float* wk_w        = (const float*)d_in[21];
  const float* wk_b        = (const float*)d_in[22];
  const float* wv_w        = (const float*)d_in[23];
  const float* wv_b        = (const float*)d_in[24];
  const float* ln1_g       = (const float*)d_in[25];
  const float* ln1_b       = (const float*)d_in[26];
  const float* ln2_g       = (const float*)d_in[27];
  const float* ln2_b       = (const float*)d_in[28];
  const float* ffn1_w      = (const float*)d_in[29];
  const float* ffn1_b      = (const float*)d_in[30];
  const float* ffn2_w      = (const float*)d_in[31];
  const float* ffn2_b      = (const float*)d_in[32];
  const float* head_fc1_w  = (const float*)d_in[33];
  const float* head_fc1_b  = (const float*)d_in[34];
  const float* head_ln_g   = (const float*)d_in[35];
  const float* head_ln_b   = (const float*)d_in[36];
  const float* head_fc2_w  = (const float*)d_in[37];
  const float* head_fc2_b  = (const float*)d_in[38];

  float* out     = (float*)d_out;
  float* preds   = out;                                   // [B,T,8]
  float* alpha0  = out + (size_t)BT_ * COUT_;             // [B,T,T]
  float* meanout = alpha0 + (size_t)B_ * 36;              // scalar

  char* wsb = (char*)d_ws;
  size_t off = 0;
  auto alloc = [&](size_t bytes) -> char* {
    char* p = wsb + off;
    off += (bytes + 255) & ~(size_t)255;
    return p;
  };
  bf16*  bt_zp = (bf16*)alloc((size_t)ZD_ * HD_ * 2);
  bf16*  bt_c1 = (bf16*)alloc((size_t)ZD_ * ZD_ * 2);
  bf16*  bt_c2 = (bf16*)alloc((size_t)ZD_ * NH_ * I_ * 2);
  bf16*  bt_wq = (bf16*)alloc((size_t)HD_ * HD_ * 2);
  bf16*  bt_wk = (bf16*)alloc((size_t)HD_ * HD_ * 2);
  bf16*  bt_wv = (bf16*)alloc((size_t)HD_ * HD_ * 2);
  bf16*  bt_f1 = (bf16*)alloc((size_t)HD_ * FF_ * 2);
  bf16*  bt_f2 = (bf16*)alloc((size_t)FF_ * HD_ * 2);
  bf16*  bt_h1 = (bf16*)alloc((size_t)T_ * HD_ * CD_ * 2);
  bf16*  z_bf  = (bf16*)alloc((size_t)B_ * ZD_ * 2);
  float* ttp1  = (float*)alloc((size_t)T_ * HD_ * 4);
  float* tpr   = (float*)alloc((size_t)T_ * HD_ * 4);
  float* Qb    = (float*)alloc((size_t)T_ * NH_ * I_ * 4);
  float* Kb    = (float*)alloc((size_t)T_ * NH_ * I_ * 4);
  bf16*  qks   = (bf16*)alloc((size_t)NH_ * 48 * I_ * 2);
  float* hb    = (float*)alloc((size_t)B_ * HD_ * 4);
  float* hbuf  = (float*)alloc((size_t)BT_ * HD_ * 4);
  bf16*  hn    = (bf16*)alloc((size_t)BT_ * HD_ * 2);
  bf16*  c1a   = (bf16*)alloc((size_t)B_ * ZD_ * 2);
  bf16*  modb  = (bf16*)alloc((size_t)B_ * NH_ * I_ * 2);
  float* edge  = (float*)alloc((size_t)B_ * NH_ * 36 * 4);
  float* qv    = (float*)alloc((size_t)BT_ * HD_ * 4);
  float* kv    = (float*)alloc((size_t)BT_ * HD_ * 4);
  float* vv    = (float*)alloc((size_t)BT_ * HD_ * 4);
  bf16*  f1a   = (bf16*)alloc((size_t)BT_ * FF_ * 2);
  float* fh    = (float*)alloc((size_t)BT_ * HD_ * 4);

  dim3 blk(256);
  dim3 gblk(128);
  auto g1 = [](long long n) { return dim3((unsigned)((n + 255) / 256)); };

  // --- weight packing (f32 [K,N] -> bf16 [N,K]) + z conversion ---
  k_pack_bt<<<g1((long long)ZD_ * HD_), blk, 0, stream>>>(tpi_zproj_w, bt_zp, ZD_, HD_);
  k_pack_bt<<<g1((long long)ZD_ * ZD_), blk, 0, stream>>>(eg_c1_w, bt_c1, ZD_, ZD_);
  k_pack_bt<<<g1((long long)ZD_ * NH_ * I_), blk, 0, stream>>>(eg_c2_w, bt_c2, ZD_, NH_ * I_);
  k_pack_bt<<<g1((long long)HD_ * HD_), blk, 0, stream>>>(wq_w, bt_wq, HD_, HD_);
  k_pack_bt<<<g1((long long)HD_ * HD_), blk, 0, stream>>>(wk_w, bt_wk, HD_, HD_);
  k_pack_bt<<<g1((long long)HD_ * HD_), blk, 0, stream>>>(wv_w, bt_wv, HD_, HD_);
  k_pack_bt<<<g1((long long)HD_ * FF_), blk, 0, stream>>>(ffn1_w, bt_f1, HD_, FF_);
  k_pack_bt<<<g1((long long)FF_ * HD_), blk, 0, stream>>>(ffn2_w, bt_f2, FF_, HD_);
  k_pack_bt3<<<g1((long long)T_ * CD_ * HD_), blk, 0, stream>>>(head_fc1_w, bt_h1, T_, CD_, HD_);
  k_cvt_bf<<<g1((long long)B_ * ZD_ / 8), blk, 0, stream>>>(z, z_bf, B_ * ZD_ / 8);

  // --- task prompt path (tiny) ---
  k_small_mm<<<g1(T_ * HD_), blk, 0, stream>>>(tt, tpi_p1_w, tpi_p1_b, ttp1, T_, 768, HD_);
  k_tpi_finish<<<T_, blk, 0, stream>>>(ttp1, tpi_ln_g, tpi_ln_b, tpi_p2_w, tpi_p2_b, tpr);
  k_small_mm<<<g1(T_ * NH_ * I_), blk, 0, stream>>>(tt, eg_q_w, eg_q_b, Qb, T_, 768, NH_ * I_);
  k_small_mm<<<g1(T_ * NH_ * I_), blk, 0, stream>>>(tt, eg_k_w, eg_k_b, Kb, T_, 768, NH_ * I_);
  k_qks<<<g1(NH_ * 48 * I_), blk, 0, stream>>>(Qb, Kb, eg_temp, qks);

  // --- big GEMMs: zproj, c1 (relu), c2 (mod), edge ---
  k_gemm4<0><<<dim3(B_ / 32, HD_ / 256), gblk, 0, stream>>>(z_bf, bt_zp, tpi_zproj_b, hb, nullptr, HD_, ZD_, ZD_);
  k_gemm4<1><<<dim3(B_ / 32, ZD_ / 256), gblk, 0, stream>>>(z_bf, bt_c1, eg_c1_b, nullptr, c1a, ZD_, ZD_, ZD_);
  k_gemm4<2><<<dim3(B_ / 32, (NH_ * I_) / 256), gblk, 0, stream>>>(c1a, bt_c2, eg_c2_b, nullptr, modb, NH_ * I_, ZD_, ZD_);
  k_gemm_edge<<<dim3(B_ / 16, 1, NH_), gblk, 0, stream>>>(modb, qks, edge);
  k_expand<<<g1((long long)BT_ * 256), blk, 0, stream>>>(hb, tpr, hbuf);

  // --- K_STEPS attention + FFN ---
  for (int step = 0; step < 2; step++) {
    k_ln_bf16<<<g1((long long)BT_ * 32), blk, 0, stream>>>(hbuf, ln1_g, ln1_b, hn, BT_);
    k_gemm4<0><<<dim3(BT_ / 32, 1), gblk, 0, stream>>>(hn, bt_wq, wq_b, qv, nullptr, HD_, HD_, HD_);
    k_gemm4<0><<<dim3(BT_ / 32, 1), gblk, 0, stream>>>(hn, bt_wk, wk_b, kv, nullptr, HD_, HD_, HD_);
    k_gemm4<0><<<dim3(BT_ / 32, 1), gblk, 0, stream>>>(hn, bt_wv, wv_b, vv, nullptr, HD_, HD_, HD_);
    k_attn<<<B_, blk, 0, stream>>>(qv, kv, vv, edge, hbuf, alpha0, step);
    k_ln_bf16<<<g1((long long)BT_ * 32), blk, 0, stream>>>(hbuf, ln2_g, ln2_b, hn, BT_);
    k_gemm4<3><<<dim3(BT_ / 32, FF_ / 256), gblk, 0, stream>>>(hn, bt_f1, ffn1_b, nullptr, f1a, FF_, HD_, HD_);
    k_gemm4<4><<<dim3(BT_ / 32, 1), gblk, 0, stream>>>(f1a, bt_f2, ffn2_b, hbuf, nullptr, HD_, FF_, FF_);
  }

  // --- heads: convert h to bf16, fc1 via WMMA (virtual concat), then LN+gelu+fc2 ---
  k_cvt_bf<<<g1((long long)BT_ * 256 / 8), blk, 0, stream>>>(hbuf, hn, BT_ * 256 / 8);
  k_gemm_head1<<<dim3(B_ / 32, 1, T_), gblk, 0, stream>>>(hn, z_bf, bt_h1, head_fc1_b, fh);
  k_head2<<<BT_, blk, 0, stream>>>(fh, head_ln_g, head_ln_b, head_fc2_w, head_fc2_b, preds);
  k_mean<<<1, blk, 0, stream>>>(alpha0, meanout);
}
